// EvolveGCNO_16106127360501
// MI455X (gfx1250) — matrix operand order repaired and checked by reading, not compiled, not run on hardware
//
#include <hip/hip_runtime.h>
#include <hip/hip_bf16.h>

// ---------------------------------------------------------------------------
// EvolveGCN-O for MI455X (gfx1250).  T=8, N=50000, E=800000, D=128.
// GEMMs run on v_wmma_f32_16x16x32_bf16 (fp32->bf16 on the fly, f32 accum).
// GCN GEMM stages pre-swizzled weights in LDS (ds_load path, padded stride).
// Edge aggregation uses relaxed agent-scope f32 atomics (L2-resident set).
// ---------------------------------------------------------------------------

#define TT 8
#define NN 50000
#define EE 800000
#define DD 128
#define STRIPS (NN / 16)          // 3125 row strips of 16
#define NDSZ ((size_t)NN * DD)    // elements per node tensor
#define LDS_SLOT 24               // bf16 per fragment-lane slot (16 used, 48B stride)

typedef __attribute__((ext_vector_type(16))) __bf16 bf16x16;
typedef __attribute__((ext_vector_type(8)))  float  f32x8;

__device__ __forceinline__ float sigmoidf_(float x) {
  return 1.0f / (1.0f + expf(-x));
}

__device__ __forceinline__ void atomic_add_relaxed(float* p, float v) {
  (void)__hip_atomic_fetch_add(p, v, __ATOMIC_RELAXED, __HIP_MEMORY_SCOPE_AGENT);
}

// A-fragment (16x32 bf16, ISA 7.12.2): lane = (half<<4)|m holds
//   elements 0..7  = A[m, kbase + 8*half + 0..7]
//   elements 8..15 = A[m, kbase + 16 + 8*half + 0..7]
__device__ __forceinline__ bf16x16 a_frag_from_f32(const float* __restrict__ row,
                                                   int kbase, int half) {
  const float* p0 = row + kbase + 8 * half;
  const float* p1 = row + kbase + 16 + 8 * half;
  bf16x16 a;
#pragma unroll
  for (int i = 0; i < 8; ++i) {
    a[i]     = (__bf16)p0[i];
    a[8 + i] = (__bf16)p1[i];
  }
  return a;
}

__device__ __forceinline__ bf16x16 a_frag_from_bf16(const __bf16* __restrict__ row,
                                                    int kbase, int half) {
  const __bf16* p0 = row + kbase + 8 * half;
  const __bf16* p1 = row + kbase + 16 + 8 * half;
  bf16x16 a;
#pragma unroll
  for (int i = 0; i < 8; ++i) {
    a[i]     = p0[i];
    a[8 + i] = p1[i];
  }
  return a;
}

// Assemble a B fragment from two 16B halves (16B-aligned source).
__device__ __forceinline__ bf16x16 b_frag_from_16B(const __bf16* p) {
  union { uint4 u[2]; bf16x16 v; } u;
  u.u[0] = *(const uint4*)(p);
  u.u[1] = *(const uint4*)(p + 8);
  return u.v;
}

// B fragments are pre-swizzled so each lane loads 16 contiguous bf16 (32B):
//   frag[(((nTile*4 + kc)*32 + lane) << 4) + idx] =
//     B[kc*32 + (idx>>3)*16 + (lane>>4)*8 + (idx&7), nTile*16 + (lane&15)]
__device__ __forceinline__ void pack_b_elem(__bf16* __restrict__ frag,
                                            int k, int n, float v) {
  int nTile = n >> 4, nn = n & 15;
  int kc = k >> 5, rem = k & 31;
  int half = (rem >> 3) & 1, idx = (rem & 7) | ((rem >> 4) << 3);
  int lane = (half << 4) | nn;
  frag[(((nTile * 4 + kc) * 32 + lane) << 4) + idx] = (__bf16)v;
}

// ---------------------------------------------------------------------------
// Utility fills
// ---------------------------------------------------------------------------
__global__ void k_fill_f32(float* __restrict__ p, float v, long n) {
  long i = (long)blockIdx.x * blockDim.x + threadIdx.x;
  if (i < n) p[i] = v;
}

// ---------------------------------------------------------------------------
// Pack a [ncol,128] fp32 weight (used as B = w^T, K=128 x ncol) into fragments
// ---------------------------------------------------------------------------
__global__ void k_pack_wT_frags(const float* __restrict__ w, __bf16* __restrict__ frag,
                                int ncol) {
  int i = blockIdx.x * 256 + threadIdx.x;         // i = n*128 + k
  if (i >= 128 * ncol) return;
  int n = i >> 7, k = i & 127;
  pack_b_elem(frag, k, n, w[(size_t)n * 128 + k]);
}

// ---------------------------------------------------------------------------
// Weight-evolution LSTM: W [128,128], h = 0 => gates = W @ wih^T + bih + bhh.
// Writes new W (fp32 state), C (in place) and pre-swizzled bf16 B-fragments.
// One block per row m; thread jj = output column.
// ---------------------------------------------------------------------------
__global__ void __launch_bounds__(128)
k_weight_lstm(const float* __restrict__ wih, const float* __restrict__ bih,
              const float* __restrict__ bhh, float* __restrict__ Wstate,
              float* __restrict__ Cstate, __bf16* __restrict__ Wfrag) {
  int m = blockIdx.x;        // 0..127
  int jj = threadIdx.x;      // 0..127
  const float* row = Wstate + (size_t)m * 128;
  float g4[4];
#pragma unroll
  for (int g = 0; g < 4; ++g) {
    const float* wr = wih + (size_t)(g * 128 + jj) * 128;
    float s = 0.f;
#pragma unroll 4
    for (int k = 0; k < 128; ++k) s += row[k] * wr[k];
    g4[g] = s + bih[g * 128 + jj] + bhh[g * 128 + jj];
  }
  __syncthreads();           // all reads of Wstate row done before overwrite
  float c  = Cstate[(size_t)m * 128 + jj];
  float c2 = sigmoidf_(g4[1]) * c + sigmoidf_(g4[0]) * tanhf(g4[2]);
  float h2 = sigmoidf_(g4[3]) * tanhf(c2);
  Cstate[(size_t)m * 128 + jj] = c2;
  Wstate[(size_t)m * 128 + jj] = h2;
  pack_b_elem(Wfrag, m, jj, h2);   // B[k=m][n=jj] for xw = x @ W
}

// ---------------------------------------------------------------------------
// GCN GEMM: Y[N,128] = X[N,128] @ W (W pre-swizzled, 8 nTiles x 4 kc).
// Weight fragments are staged once per block into LDS (48B/lane-slot padding
// -> 2-way worst-case bank conflicts) and fed to WMMA via ds_load_b128.
// 8 waves / block, 1 wave = one 16-row strip, loops 8 column tiles.
// ---------------------------------------------------------------------------
__global__ void __launch_bounds__(256)
k_gemm_gcn(const float* __restrict__ X, const __bf16* __restrict__ Wfrag,
           float* __restrict__ Y, int nrows) {
  __shared__ __attribute__((aligned(16))) __bf16 lW[1024 * LDS_SLOT];  // 48 KB

  // cooperative stage: 1024 fragment-lane slots of 32B each
  for (int s = threadIdx.x; s < 1024; s += 256) {
    const uint4* g = (const uint4*)(Wfrag + (s << 4));
    uint4* d = (uint4*)(lW + s * LDS_SLOT);
    d[0] = g[0];
    d[1] = g[1];
  }
  __syncthreads();   // all threads reach this before any wave may exit

  int wave = threadIdx.x >> 5;
  int lane = threadIdx.x & 31;
  int strip = blockIdx.x * 8 + wave;
  if (strip * 16 >= nrows) return;

  int lm = lane & 15, half = lane >> 4;
  const float* row = X + (size_t)(strip * 16 + lm) * 128;
  bf16x16 a[4];
#pragma unroll
  for (int kc = 0; kc < 4; ++kc) a[kc] = a_frag_from_f32(row, kc * 32, half);
#pragma unroll
  for (int nt = 0; nt < 8; ++nt) {
    f32x8 acc = {};
#pragma unroll
    for (int kc = 0; kc < 4; ++kc) {
      bf16x16 b = b_frag_from_16B(lW + ((nt * 4 + kc) * 32 + lane) * LDS_SLOT);
      acc = __builtin_amdgcn_wmma_f32_16x16x32_bf16(false, a[kc], false, b,
                                                    (short)0, acc, false, false);
    }
    float* out = Y + (size_t)(strip * 16 + 8 * half) * 128 + nt * 16 + lm;
#pragma unroll
    for (int j = 0; j < 8; ++j) out[(size_t)j * 128] = acc[j];
  }
}

// ---------------------------------------------------------------------------
// Degree / aggregation kernels
// ---------------------------------------------------------------------------
__global__ void k_deg(const int* __restrict__ dst, float* __restrict__ deg, int E) {
  int e = blockIdx.x * 256 + threadIdx.x;
  if (e < E) atomic_add_relaxed(&deg[dst[e]], 1.0f);
}

__global__ void k_dinv(const float* __restrict__ deg, float* __restrict__ dinv, int n) {
  int i = blockIdx.x * 256 + threadIdx.x;
  if (i < n) dinv[i] = rsqrtf(deg[i]);   // deg >= 1 (self loop)
}

// out[i][j] = xw[i][j]*dinv[i]^2 + bias[j]   (self loop + bias)
__global__ void k_selfloop_init(float* __restrict__ out, const float* __restrict__ xw,
                                const float* __restrict__ dinv,
                                const float* __restrict__ bias, long n) {
  long i = (long)blockIdx.x * 256 + threadIdx.x;
  if (i >= n) return;
  int node = (int)(i >> 7), j = (int)(i & 127);
  float di = dinv[node];
  out[i] = xw[i] * di * di + bias[j];
}

// out[d] += xw[s] * dinv[s]*dinv[d] ; 2 edges per 256-thread block
__global__ void __launch_bounds__(256)
k_edge_scatter(const int* __restrict__ src, const int* __restrict__ dst,
               const float* __restrict__ xw, const float* __restrict__ dinv,
               float* __restrict__ out, int E) {
  int e = blockIdx.x * 2 + (threadIdx.x >> 7);
  int j = threadIdx.x & 127;
  if (e >= E) return;
  int s = src[e], d = dst[e];
  float w = dinv[s] * dinv[d];
  atomic_add_relaxed(&out[(size_t)d * 128 + j], xw[(size_t)s * 128 + j] * w);
}

// ---------------------------------------------------------------------------
// BatchNorm (axis 0) + ReLU
// ---------------------------------------------------------------------------
__global__ void __launch_bounds__(128)
k_bn_stats(const float* __restrict__ z, float* __restrict__ sum,
           float* __restrict__ sumsq, int nrows) {
  int j = threadIdx.x;
  int r0 = blockIdx.x * 256;
  int rend = r0 + 256 < nrows ? r0 + 256 : nrows;
  float s = 0.f, sq = 0.f;
  for (int r = r0; r < rend; ++r) {
    float v = z[(size_t)r * 128 + j];
    s += v; sq += v * v;
  }
  atomic_add_relaxed(&sum[j], s);
  atomic_add_relaxed(&sumsq[j], sq);
}

__global__ void k_bn_apply(float* __restrict__ z, const float* __restrict__ sum,
                           const float* __restrict__ sumsq,
                           const float* __restrict__ bn_g,
                           const float* __restrict__ bn_b, long n, float invN) {
  long i = (long)blockIdx.x * 256 + threadIdx.x;
  if (i >= n) return;
  int j = (int)(i & 127);
  float mu  = sum[j] * invN;
  float var = sumsq[j] * invN - mu * mu;
  float v = (z[i] - mu) * rsqrtf(var + 1e-5f) * bn_g[j] + bn_b[j];
  z[i] = v > 0.f ? v : 0.f;
}

// ---------------------------------------------------------------------------
// Fused node-LSTM step: gates = X @ wih^T (+ Hbf16 @ whh^T) + bih + bhh,
// then the LSTM cell applied in-registers. One block per 16-row strip; wave
// jt handles output columns [jt*16, jt*16+16) for all 4 gates (32 WMMAs).
// Per-wave B tiles are disjoint, so they stream from global (L2-resident).
// ---------------------------------------------------------------------------
__global__ void __launch_bounds__(256)
k_lstm_fused(const float* __restrict__ Xf32, const __bf16* __restrict__ Hbf16,
             int use_h, const float* __restrict__ Cin,
             const __bf16* __restrict__ WihFrag, const __bf16* __restrict__ WhhFrag,
             const float* __restrict__ bih, const float* __restrict__ bhh,
             float* __restrict__ Hout, float* __restrict__ Cout,
             __bf16* __restrict__ HbfOut) {
  int jt = threadIdx.x >> 5;    // 0..7 column tile within D
  int lane = threadIdx.x & 31;
  int strip = blockIdx.x;       // 0..3124
  int lm = lane & 15, half = lane >> 4;

  const float* xrow = Xf32 + (size_t)(strip * 16 + lm) * 128;
  bf16x16 ax[4], ah[4];
#pragma unroll
  for (int kc = 0; kc < 4; ++kc) ax[kc] = a_frag_from_f32(xrow, kc * 32, half);
  if (use_h) {
    const __bf16* hrow = Hbf16 + (size_t)(strip * 16 + lm) * 128;
#pragma unroll
    for (int kc = 0; kc < 4; ++kc) ah[kc] = a_frag_from_bf16(hrow, kc * 32, half);
  }
  __syncthreads();   // Hbf16 rows fully read before any wave overwrites HbfOut

  f32x8 acc[4];
#pragma unroll
  for (int g = 0; g < 4; ++g) {
    f32x8 c = {};
    int nt = g * 8 + jt;        // gate section g occupies column tiles g*8..g*8+7
#pragma unroll
    for (int kc = 0; kc < 4; ++kc) {
      bf16x16 b = *(const bf16x16*)(WihFrag + (((nt * 4 + kc) * 32 + lane) << 4));
      c = __builtin_amdgcn_wmma_f32_16x16x32_bf16(false, ax[kc], false, b,
                                                  (short)0, c, false, false);
    }
    if (use_h) {
#pragma unroll
      for (int kc = 0; kc < 4; ++kc) {
        bf16x16 b = *(const bf16x16*)(WhhFrag + (((nt * 4 + kc) * 32 + lane) << 4));
        c = __builtin_amdgcn_wmma_f32_16x16x32_bf16(false, ah[kc], false, b,
                                                    (short)0, c, false, false);
      }
    }
    acc[g] = c;
  }

  int col = jt * 16 + lm;       // column within D
#pragma unroll
  for (int j = 0; j < 8; ++j) {
    size_t idx = (size_t)(strip * 16 + 8 * half + j) * 128 + col;
    float gi = acc[0][j] + bih[col]       + bhh[col];
    float gf = acc[1][j] + bih[128 + col] + bhh[128 + col];
    float gg = acc[2][j] + bih[256 + col] + bhh[256 + col];
    float go = acc[3][j] + bih[384 + col] + bhh[384 + col];
    float c2 = sigmoidf_(gf) * Cin[idx] + sigmoidf_(gi) * tanhf(gg);
    float h2 = sigmoidf_(go) * tanhf(c2);
    Cout[idx] = c2;
    Hout[idx] = h2;
    if (HbfOut) HbfOut[idx] = (__bf16)h2;
  }
}

// ---------------------------------------------------------------------------
// Host orchestration
// ---------------------------------------------------------------------------
extern "C" void kernel_launch(void* const* d_in, const int* in_sizes, int n_in,
                              void* d_out, int out_size, void* d_ws, size_t ws_size,
                              hipStream_t stream) {
  const float* x_seq   = (const float*)d_in[0];
  const int*   ei_seq  = (const int*)d_in[1];
  const float* conv_w0 = (const float*)d_in[2];
  const float* conv_b0 = (const float*)d_in[3];
  const float* conv_w1 = (const float*)d_in[4];
  const float* conv_b1 = (const float*)d_in[5];
  const float* bn_g    = (const float*)d_in[6];
  const float* bn_b    = (const float*)d_in[7];
  const float* l0_wih  = (const float*)d_in[8];
  const float* l0_bih  = (const float*)d_in[10];
  const float* l0_bhh  = (const float*)d_in[11];
  const float* l1_wih  = (const float*)d_in[12];
  const float* l1_bih  = (const float*)d_in[14];
  const float* l1_bhh  = (const float*)d_in[15];
  const float* fl_wih  = (const float*)d_in[16];
  const float* fl_whh  = (const float*)d_in[17];
  const float* fl_bih  = (const float*)d_in[18];
  const float* fl_bhh  = (const float*)d_in[19];
  float* out = (float*)d_out;

  // workspace carve-up (256B aligned)
  char* base = (char*)d_ws;
  size_t off = 0;
  auto carve = [&](size_t bytes) { char* p = base + off; off += (bytes + 255) & ~(size_t)255; return p; };
  float*  W0s  = (float*)carve(128 * 128 * 4);
  float*  c0s  = (float*)carve(128 * 128 * 4);
  float*  W1s  = (float*)carve(128 * 128 * 4);
  float*  c1s  = (float*)carve(128 * 128 * 4);
  __bf16* W0f  = (__bf16*)carve(128 * 128 * 2);
  __bf16* W1f  = (__bf16*)carve(128 * 128 * 2);
  __bf16* wihF = (__bf16*)carve(128 * 512 * 2);
  __bf16* whhF = (__bf16*)carve(128 * 512 * 2);
  float*  deg  = (float*)carve((size_t)NN * 4);
  float*  dinv = (float*)carve((size_t)NN * 4);
  float*  bnsum = (float*)carve(512);
  float*  bnsq  = (float*)carve(512);
  float*  xw   = (float*)carve(NDSZ * 4);
  float*  zbuf = (float*)carve(NDSZ * 4);
  float*  hbuf = (float*)carve(NDSZ * 4);
  float*  cbuf = (float*)carve(NDSZ * 4);
  __bf16* hbf  = (__bf16*)carve(NDSZ * 2);
  (void)ws_size; (void)n_in; (void)in_sizes; (void)out_size;

  const long nd = (long)NDSZ;
  const int blkND = (int)((nd + 255) / 256);     // 25000
  const int blkN  = (NN + 255) / 256;            // 196
  const int blkE  = (EE + 255) / 256;            // 3125
  const int blkGemm = (STRIPS + 7) / 8;          // 391

  // ---- init state -------------------------------------------------------
  hipMemcpyAsync(W0s, conv_w0, 128 * 128 * 4, hipMemcpyDeviceToDevice, stream);
  hipMemcpyAsync(W1s, conv_w1, 128 * 128 * 4, hipMemcpyDeviceToDevice, stream);
  k_fill_f32<<<64, 256, 0, stream>>>(c0s, 0.f, 128 * 128);
  k_fill_f32<<<64, 256, 0, stream>>>(c1s, 0.f, 128 * 128);
  k_fill_f32<<<blkND, 256, 0, stream>>>(hbuf, 0.f, nd);
  k_fill_f32<<<blkND, 256, 0, stream>>>(cbuf, 0.f, nd);
  k_fill_f32<<<(int)((nd / 2 + 255) / 256), 256, 0, stream>>>((float*)hbf, 0.f, nd / 2);
  k_pack_wT_frags<<<256, 256, 0, stream>>>(fl_wih, wihF, 512);
  k_pack_wT_frags<<<256, 256, 0, stream>>>(fl_whh, whhF, 512);

  // ---- GCN scan (timestep 7 is dead: output = zs[:-1] ++ z_future) ------
  for (int t = 0; t < TT - 1; ++t) {
    const float* xt  = x_seq + (size_t)t * NDSZ;
    const int*   src = ei_seq + (size_t)t * 2 * EE;
    const int*   dst = src + EE;
    float* out_t = out + (size_t)t * NDSZ;

    // degree / normalization shared by both convs of this timestep
    k_fill_f32<<<blkN, 256, 0, stream>>>(deg, 1.0f, NN);   // self loop
    k_deg<<<blkE, 256, 0, stream>>>(dst, deg, EE);
    k_dinv<<<blkN, 256, 0, stream>>>(deg, dinv, NN);

    // conv 1
    k_weight_lstm<<<128, 128, 0, stream>>>(l0_wih, l0_bih, l0_bhh, W0s, c0s, W0f);
    k_gemm_gcn<<<blkGemm, 256, 0, stream>>>(xt, W0f, xw, NN);
    k_selfloop_init<<<blkND, 256, 0, stream>>>(zbuf, xw, dinv, conv_b0, nd);
    k_edge_scatter<<<EE / 2, 256, 0, stream>>>(src, dst, xw, dinv, zbuf, EE);

    // batchnorm + relu
    k_fill_f32<<<1, 256, 0, stream>>>(bnsum, 0.f, 128);
    k_fill_f32<<<1, 256, 0, stream>>>(bnsq, 0.f, 128);
    k_bn_stats<<<blkN, 128, 0, stream>>>(zbuf, bnsum, bnsq, NN);
    k_bn_apply<<<blkND, 256, 0, stream>>>(zbuf, bnsum, bnsq, bn_g, bn_b, nd,
                                          1.0f / (float)NN);

    // conv 2 -> output slice t
    k_weight_lstm<<<128, 128, 0, stream>>>(l1_wih, l1_bih, l1_bhh, W1s, c1s, W1f);
    k_gemm_gcn<<<blkGemm, 256, 0, stream>>>(zbuf, W1f, xw, NN);
    k_selfloop_init<<<blkND, 256, 0, stream>>>(out_t, xw, dinv, conv_b1, nd);
    k_edge_scatter<<<EE / 2, 256, 0, stream>>>(src, dst, xw, dinv, out_t, EE);
  }

  // ---- future-prediction LSTM over zs[0..6] -----------------------------
  for (int s = 0; s < TT - 1; ++s) {
    k_lstm_fused<<<STRIPS, 256, 0, stream>>>(
        out + (size_t)s * NDSZ, hbf, /*use_h=*/(s > 0), cbuf,
        wihF, whhF, fl_bih, fl_bhh, hbuf, cbuf, hbf);
  }
  // final step: x = h_last, h reset to 0, c = c_last -> out[7]
  k_lstm_fused<<<STRIPS, 256, 0, stream>>>(
      hbuf, hbf, /*use_h=*/0, cbuf, wihF, whhF, fl_bih, fl_bhh,
      out + (size_t)7 * NDSZ, cbuf, (__bf16*)nullptr);
}
